// MoGRUCell_87351044866265
// MI455X (gfx1250) — matrix operand order repaired and checked by reading, not compiled, not run on hardware
//
#include <hip/hip_runtime.h>
#include <hip/hip_bf16.h>

typedef __attribute__((ext_vector_type(16))) __bf16 v16bf;
typedef __attribute__((ext_vector_type(8)))  float  v8f;
typedef unsigned short bfraw;
typedef unsigned int   u32t;

__device__ __forceinline__ bfraw f2bf(float f) {
    u32t u = __float_as_uint(f);
    u = (u + 0x7FFFu + ((u >> 16) & 1u)) >> 16;
    return (bfraw)u;
}
__device__ __forceinline__ float sigm(float x) { return 1.0f / (1.0f + __expf(-x)); }

// ---------------- conversion kernels ----------------
__global__ void mogru_cvt_bf16(const float* __restrict__ s, bfraw* __restrict__ d, int n) {
    for (int i = blockIdx.x * blockDim.x + threadIdx.x; i < n; i += gridDim.x * blockDim.x)
        d[i] = f2bf(s[i]);
}

// W1 = [W_ru_w ; W_beta_w]  -> (3072, 2048) bf16
__global__ void mogru_cvt_w1(const float* __restrict__ wru, const float* __restrict__ wbeta,
                             bfraw* __restrict__ d) {
    const int n = 3072 * 2048;
    for (int i = blockIdx.x * blockDim.x + threadIdx.x; i < n; i += gridDim.x * blockDim.x) {
        int row = i >> 11, k = i & 2047;
        float v = (row < 2048) ? wru[(size_t)row * 2048 + k]
                               : wbeta[(size_t)(row - 2048) * 2048 + k];
        d[i] = f2bf(v);
    }
}

// W2[n][k] = (k<1024) ? W_h_w[n][k] : U_h_w[n][k-1024]  -> (1024, 2048) bf16
__global__ void mogru_cvt_w2(const float* __restrict__ wh, const float* __restrict__ uh,
                             bfraw* __restrict__ d) {
    const int n = 1024 * 2048;
    for (int i = blockIdx.x * blockDim.x + threadIdx.x; i < n; i += gridDim.x * blockDim.x) {
        int row = i >> 11, k = i & 2047;
        float v = (k < 1024) ? wh[(size_t)row * 1024 + k]
                             : uh[(size_t)row * 1024 + (k - 1024)];
        d[i] = f2bf(v);
    }
}

// ---------------- WMMA GEMM with async Global->LDS double-buffered pipeline ----------
// C[m][n] = sum_k A[m][k] * W[n][k],  A = [Alo | Ahi] (each 8192x1024 bf16), W: (N x 2048) bf16
// Block tile 128x128, 8 waves (2x4), wave tile 64x32, K-step 32.
union Frag16 { uint4 q[2]; v16bf v; };

template <int MODE>
__global__ __launch_bounds__(256) void mogru_gemm_wmma(
    const bfraw* __restrict__ Alo, const bfraw* __restrict__ Ahi,
    const bfraw* __restrict__ W,
    const float* __restrict__ bias0, const float* __restrict__ bias1,
    const float* __restrict__ hprev,
    bfraw* __restrict__ rh_out, float* __restrict__ u_out,
    float* __restrict__ beta_out, float* __restrict__ z_out)
{
    constexpr int LDT = 40;                 // padded LDS row stride (elements)
    __shared__ bfraw sA[2][128 * LDT];
    __shared__ bfraw sB[2][128 * LDT];

    const int tid  = threadIdx.x;
    const int lane = tid & 31;
    const int wid  = tid >> 5;
    const int wm   = wid >> 2;              // 0..1 -> M offset wm*64
    const int wn   = wid & 3;               // 0..3 -> N offset wn*32
    const int m0   = blockIdx.y * 128;
    const int n0   = blockIdx.x * 128;

    // staging mapping: 512 16B-chunks per tile; thread covers rows srow and srow+64
    const int srow = tid >> 2;
    const int ssub = (tid & 3) * 8;         // element offset within row

    // issue one K-tile's async DMA (4 x global_load_async_to_lds_b128 per thread)
    auto issue = [&](int buf, int k0) {
        const bfraw* Asrc = (k0 < 1024) ? Alo : Ahi;
        const int ka = k0 & 1023;
        const bfraw* ga0 = &Asrc[(size_t)(m0 + srow) * 1024 + ka + ssub];
        const bfraw* ga1 = &Asrc[(size_t)(m0 + srow + 64) * 1024 + ka + ssub];
        const bfraw* gb0 = &W[(size_t)(n0 + srow) * 2048 + k0 + ssub];
        const bfraw* gb1 = &W[(size_t)(n0 + srow + 64) * 2048 + k0 + ssub];
        u32t la0 = (u32t)(size_t)&sA[buf][srow * LDT + ssub];
        u32t la1 = (u32t)(size_t)&sA[buf][(srow + 64) * LDT + ssub];
        u32t lb0 = (u32t)(size_t)&sB[buf][srow * LDT + ssub];
        u32t lb1 = (u32t)(size_t)&sB[buf][(srow + 64) * LDT + ssub];
        asm volatile(
            "global_load_async_to_lds_b128 %0, %4, off\n\t"
            "global_load_async_to_lds_b128 %1, %5, off\n\t"
            "global_load_async_to_lds_b128 %2, %6, off\n\t"
            "global_load_async_to_lds_b128 %3, %7, off"
            :
            : "v"(la0), "v"(la1), "v"(lb0), "v"(lb1),
              "v"(ga0), "v"(ga1), "v"(gb0), "v"(gb1)
            : "memory");
    };

    const v8f vzero = {0.f, 0.f, 0.f, 0.f, 0.f, 0.f, 0.f, 0.f};
    v8f acc[4][2];
#pragma unroll
    for (int i = 0; i < 4; ++i)
#pragma unroll
        for (int j = 0; j < 2; ++j) acc[i][j] = vzero;

    const int ar  = lane & 15;
    const int akb = (lane < 16) ? 0 : 8;    // A lane K-base (ISA 16-bit A layout)
    const int bkb = (lane < 16) ? 0 : 8;    // B lane K-base (same layout as A)

    issue(0, 0);                            // prologue: tile 0 -> buffer 0

    for (int t = 0; t < 64; ++t) {
        const int buf = t & 1;

        // all waves finished reading buf^1 (tile t-1, read last iteration) before we overwrite it
        __syncthreads();
        if (t + 1 < 64) {
            issue(buf ^ 1, (t + 1) * 32);
            // retire tile t's 4 DMAs (in-order), leave tile t+1's 4 in flight
            asm volatile("s_wait_asynccnt 4" ::: "memory");
        } else {
            asm volatile("s_wait_asynccnt 0" ::: "memory");
        }
        // make every wave's DMA for tile t visible workgroup-wide
        __syncthreads();

        Frag16 a[4], b[2];
#pragma unroll
        for (int mi = 0; mi < 4; ++mi) {
            const bfraw* p = &sA[buf][(wm * 64 + mi * 16 + ar) * LDT + akb];
            a[mi].q[0] = *(const uint4*)p;          // K = kb .. kb+7
            a[mi].q[1] = *(const uint4*)(p + 16);   // K = kb+16 .. kb+23
        }
#pragma unroll
        for (int ni = 0; ni < 2; ++ni) {
            const bfraw* p = &sB[buf][(wn * 32 + ni * 16 + ar) * LDT + bkb];
            b[ni].q[0] = *(const uint4*)p;          // K = kb .. kb+7
            b[ni].q[1] = *(const uint4*)(p + 16);   // K = kb+16 .. kb+23
        }
#pragma unroll
        for (int mi = 0; mi < 4; ++mi)
#pragma unroll
            for (int ni = 0; ni < 2; ++ni)
                acc[mi][ni] = __builtin_amdgcn_wmma_f32_16x16x32_bf16(
                    false, a[mi].v, false, b[ni].v, (short)0, acc[mi][ni], false, false);
    }

    // ---------------- epilogue ----------------
    const int half = lane >> 4;             // C layout: VGPR r -> M = r + half*8
#pragma unroll
    for (int mi = 0; mi < 4; ++mi) {
#pragma unroll
        for (int ni = 0; ni < 2; ++ni) {
            const int col = n0 + wn * 32 + ni * 16 + ar;
#pragma unroll
            for (int r = 0; r < 8; ++r) {
                const int row = m0 + wm * 64 + mi * 16 + half * 8 + r;
                float v = acc[mi][ni][r];
                if (MODE == 0) {
                    if (col < 1024) {                               // r gate -> r*h_prev (bf16)
                        float rg = sigm(v + bias0[col]);
                        float rh = rg * hprev[(size_t)row * 1024 + col];
                        rh_out[(size_t)row * 1024 + col] = f2bf(rh);
                    } else if (col < 2048) {                        // u gate (f32)
                        u_out[(size_t)row * 1024 + (col - 1024)] = sigm(v + bias0[col]);
                    } else {                                        // beta -> d_out slice 3
                        beta_out[(size_t)row * 1024 + (col - 2048)] = sigm(v + bias1[col - 2048]);
                    }
                } else {
                    z_out[(size_t)row * 1024 + col] = v + bias0[col];
                }
            }
        }
    }
}

// ---------------- finalize: tanh, v_t, h_t, LayerNorm ----------------
__global__ __launch_bounds__(256) void mogru_finalize(
    const float* __restrict__ hprev, const float* __restrict__ vprev,
    const float* __restrict__ ubuf, const float* __restrict__ zbuf,
    const float* __restrict__ beta, const float* __restrict__ lnw,
    const float* __restrict__ lnb, float* __restrict__ hout, float* __restrict__ vout)
{
    const int row = blockIdx.x;
    const size_t base = (size_t)row * 1024;
    const int tid = threadIdx.x;

    float htv[4];
    float s = 0.f, q = 0.f;
#pragma unroll
    for (int i = 0; i < 4; ++i) {
        const int j = tid + i * 256;
        float hp = hprev[base + j];
        float u  = ubuf[base + j];
        float bt = beta[base + j];
        float ht = tanhf(zbuf[base + j]);
        float dd = ht - hp;
        float v  = bt * vprev[base + j] + (1.f - bt) * dd;
        vout[base + j] = v;
        float h = hp + u * v;
        htv[i] = h;
        s += h;
        q += h * h;
    }
#pragma unroll
    for (int o = 16; o > 0; o >>= 1) {
        s += __shfl_xor(s, o, 32);
        q += __shfl_xor(q, o, 32);
    }
    __shared__ float rs[8], rq[8];
    const int lane = tid & 31, wid = tid >> 5;
    if (lane == 0) { rs[wid] = s; rq[wid] = q; }
    __syncthreads();
    if (tid == 0) {
        float ts = 0.f, tq = 0.f;
        for (int i = 0; i < 8; ++i) { ts += rs[i]; tq += rq[i]; }
        rs[0] = ts; rq[0] = tq;
    }
    __syncthreads();
    const float mu   = rs[0] * (1.f / 1024.f);
    const float var  = rq[0] * (1.f / 1024.f) - mu * mu;
    const float rstd = rsqrtf(var + 1e-5f);
#pragma unroll
    for (int i = 0; i < 4; ++i) {
        const int j = tid + i * 256;
        hout[base + j] = (htv[i] - mu) * rstd * lnw[j] + lnb[j];
    }
}

// ---------------- host launcher ----------------
extern "C" void kernel_launch(void* const* d_in, const int* in_sizes, int n_in,
                              void* d_out, int out_size, void* d_ws, size_t ws_size,
                              hipStream_t stream) {
    const float* x_t      = (const float*)d_in[0];
    const float* h_prev   = (const float*)d_in[1];
    const float* v_prev   = (const float*)d_in[2];
    const float* W_ru_w   = (const float*)d_in[3];
    const float* W_ru_b   = (const float*)d_in[4];
    const float* W_beta_w = (const float*)d_in[5];
    const float* W_beta_b = (const float*)d_in[6];
    const float* W_h_w    = (const float*)d_in[7];
    const float* W_h_b    = (const float*)d_in[8];
    const float* U_h_w    = (const float*)d_in[9];
    const float* ln_w     = (const float*)d_in[10];
    const float* ln_b     = (const float*)d_in[11];

    char* ws = (char*)d_ws;
    const size_t MB = 1024 * 1024;
    bfraw* xbf  = (bfraw*)(ws);                         // 16 MB
    bfraw* hbf  = (bfraw*)(ws + 16 * MB);               // 16 MB
    bfraw* rhbf = (bfraw*)(ws + 32 * MB);               // 16 MB
    bfraw* w1bf = (bfraw*)(ws + 48 * MB);               // 12 MB
    bfraw* w2bf = (bfraw*)(ws + 48 * MB + 12582912);    //  4 MB
    float* ubuf = (float*)(ws + 64 * MB);               // 32 MB
    float* zbuf = (float*)(ws + 96 * MB);               // 32 MB (total 128 MB)

    float* h_out = (float*)d_out;
    float* v_out = h_out + (size_t)8192 * 1024;
    float* b_out = v_out + (size_t)8192 * 1024;

    mogru_cvt_bf16<<<8192, 256, 0, stream>>>(x_t, xbf, 8192 * 1024);
    mogru_cvt_bf16<<<8192, 256, 0, stream>>>(h_prev, hbf, 8192 * 1024);
    mogru_cvt_w1<<<6144, 256, 0, stream>>>(W_ru_w, W_beta_w, w1bf);
    mogru_cvt_w2<<<2048, 256, 0, stream>>>(W_h_w, U_h_w, w2bf);

    // GEMM1: (8192 x 3072 x 2048): ru + beta, epilogue builds r*h (bf16), u, beta
    mogru_gemm_wmma<0><<<dim3(24, 64), 256, 0, stream>>>(
        xbf, hbf, w1bf, W_ru_b, W_beta_b, h_prev, rhbf, ubuf, b_out, nullptr);

    // GEMM2: (8192 x 1024 x 2048): [x | r*h] @ [W_h | U_h]^T + b_h
    mogru_gemm_wmma<1><<<dim3(8, 64), 256, 0, stream>>>(
        xbf, rhbf, w2bf, W_h_b, nullptr, nullptr, nullptr, nullptr, nullptr, zbuf);

    mogru_finalize<<<8192, 256, 0, stream>>>(h_prev, v_prev, ubuf, zbuf, b_out,
                                             ln_w, ln_b, h_out, v_out);
}
